// Registration_5720896438661
// MI455X (gfx1250) — matrix operand order repaired and checked by reading, not compiled, not run on hardware
//
#include <hip/hip_runtime.h>
#include <hip/hip_bf16.h>

typedef __bf16 bf16_t;
typedef __attribute__((ext_vector_type(16))) __bf16 v16bf;
typedef __attribute__((ext_vector_type(8)))  __bf16 v8bf;
typedef __attribute__((ext_vector_type(8)))  float  v8f;

// ---------------------------------------------------------------------------
// Layout conversion: fp32 NCHW -> bf16 NHWC (with channel offset for concat)
// ---------------------------------------------------------------------------
__global__ void nchw_to_nhwc_bf16(const float* __restrict__ src, bf16_t* __restrict__ dst,
                                  int Bn, int C, int H, int W, int Ctot, int c0) {
    size_t i = (size_t)blockIdx.x * 256 + threadIdx.x;
    size_t total = (size_t)Bn * C * H * W;
    if (i >= total) return;
    int w = (int)(i % W); size_t t = i / W;
    int h = (int)(t % H); t /= H;
    int c = (int)(t % C); int b = (int)(t / C);
    dst[((size_t)(b * H + h) * W + w) * Ctot + c0 + c] = (bf16_t)src[i];
}

// ---------------------------------------------------------------------------
// Weight prepack into WMMA B-fragment layout, bf16.
// Packed index = (((tap*nq + q)*nbc + nb)*32 + lane)*16 + e
//   n = nb*16 + (lane&15)        (output channel, zero-padded to nbc*16)
//   k = q*32 + (e>>3)*16 + (lane>>4)*8 + (e&7)  (input channel)
// mode 0: conv/deform weights  w[Cout][Cin][ks][ks]
// mode 1: conv-transpose       w[Cin ][Cout][ks][ks] (direct gather form, no flip)
// ---------------------------------------------------------------------------
__global__ void prepack_w(const float* __restrict__ src, bf16_t* __restrict__ dst,
                          int Cout, int Cin, int ks, int nq, int nbc, int mode) {
    int tid = blockIdx.x * 256 + threadIdx.x;
    int total = ks * ks * nq * nbc * 512;
    if (tid >= total) return;
    int e    = tid & 15;
    int lane = (tid >> 4) & 31;
    int tile = tid >> 9;
    int nb  = tile % nbc;
    int t2  = tile / nbc;
    int q   = t2 % nq;
    int tap = t2 / nq;
    int kr = tap / ks, kc = tap % ks;
    int k = q * 32 + (e >> 3) * 16 + (lane >> 4) * 8 + (e & 7);
    int n = nb * 16 + (lane & 15);
    float v = 0.f;
    if (n < Cout && k < Cin) {
        if (mode == 0) v = src[(((size_t)n * Cin + k) * ks + kr) * ks + kc];
        else           v = src[(((size_t)k * Cout + n) * ks + kr) * ks + kc];
    }
    dst[tid] = (bf16_t)v;
}

// ---------------------------------------------------------------------------
// Generic conv (KS x KS, stride, pad) as tap-wise implicit GEMM on WMMA.
// One wave = 16 output pixels x (NBLK*16) output channels: A fragment loaded
// once per (tap, chunk), reused by NBLK back-to-back WMMAs (ILP + traffic/4).
// All tap/chunk/N-block loops unrolled -> constant load offsets.
// OUT_MODE: 0 = bf16 NHWC, 1 = fp32 NHWC (CoutStore ch), 2 = fp32 NCHW
// ---------------------------------------------------------------------------
template<int OUT_MODE, int NQ, int NBLK, int KS, int NBC>
__global__ __launch_bounds__(32)
void conv_wmma(const bf16_t* __restrict__ in, const bf16_t* __restrict__ wp,
               const float* __restrict__ bias, void* __restrict__ outv,
               int Hin, int Win, int Ho, int Wo,
               int CoutReal, int CoutStore, int stride, int pad) {
    constexpr int Cin = NQ * 32;
    constexpr int ZB  = NBC / NBLK;
    int lane = threadIdx.x;
    int m  = lane & 15;     // pixel within tile (A-matrix M row)
    int hv = lane >> 4;     // half-wave selector (K split)
    int w0 = blockIdx.x * 16;
    int ho = blockIdx.y;
    int b   = blockIdx.z / ZB;
    int nb0 = (blockIdx.z % ZB) * NBLK;

    const bf16_t* wlane = wp + (size_t)nb0 * 512 + lane * 16;

    v8f acc[NBLK] = {};
#pragma unroll
    for (int tap = 0; tap < KS * KS; ++tap) {
        int kr = tap / KS, kc = tap % KS;
        int hi = ho * stride + kr - pad;
        if (hi >= 0 && hi < Hin) {                    // wave-uniform
            int wi = (w0 + m) * stride + kc - pad;    // per-lane
            bool vld = (wi >= 0) && (wi < Win);
            const bf16_t* ap = in + ((size_t)(b * Hin + hi) * Win + (vld ? wi : 0)) * Cin + hv * 8;
#pragma unroll
            for (int q = 0; q < NQ; ++q) {
                v8bf alo = {}, ahi = {};
                if (vld) {
                    alo = *(const v8bf*)(ap + q * 32);        // K = k0 .. k0+7
                    ahi = *(const v8bf*)(ap + q * 32 + 16);   // K = k0+16 .. k0+23
                }
                v16bf a = __builtin_shufflevector(alo, ahi,
                            0,1,2,3,4,5,6,7,8,9,10,11,12,13,14,15);
#pragma unroll
                for (int j = 0; j < NBLK; ++j) {
                    v16bf bb = *(const v16bf*)(wlane + ((tap * NQ + q) * NBC + j) * 512);
                    acc[j] = __builtin_amdgcn_wmma_f32_16x16x32_bf16(
                                false, a, false, bb, (short)0, acc[j], false, false);
                }
            }
        }
    }

#pragma unroll
    for (int j = 0; j < NBLK; ++j) {
        int n = (nb0 + j) * 16 + (lane & 15);
        float bz = (n < CoutReal) ? bias[n] : 0.f;
#pragma unroll
        for (int r = 0; r < 8; ++r) {
            int pix = w0 + hv * 8 + r;                // D row M = r + 8*hv
            float v = acc[j][r] + bz;
            if (OUT_MODE == 0) {
                ((bf16_t*)outv)[((size_t)(b * Ho + ho) * Wo + pix) * CoutStore + n] = (bf16_t)v;
            } else if (OUT_MODE == 1) {
                ((float*)outv)[((size_t)(b * Ho + ho) * Wo + pix) * CoutStore + n] = v;
            } else {
                ((float*)outv)[(((size_t)b * CoutStore + n) * Ho + ho) * Wo + pix] = v;
            }
        }
    }
}

// ---------------------------------------------------------------------------
// 4x4 stride-2 transposed conv: fixed output parity -> 4 wave-uniform taps,
// each a K=Cin WMMA GEMM. Tile = 16 same-parity output pixels x 64 channels.
// ---------------------------------------------------------------------------
__global__ __launch_bounds__(32)
void convt_wmma(const bf16_t* __restrict__ in, const bf16_t* __restrict__ wp,
                const float* __restrict__ bias, bf16_t* __restrict__ out,
                int Hin, int Win, int Ho, int Wo) {
    constexpr int NQ = 2, NBLK = 4, NBC = 4, Cin = 64, Cout = 64;
    int lane = threadIdx.x;
    int m  = lane & 15;
    int hv = lane >> 4;
    int b   = blockIdx.z >> 1;
    int par = blockIdx.z & 1;
    int ho = blockIdx.y;
    int wbase = blockIdx.x * 32 + par;
    int wo = wbase + 2 * m;

    const bf16_t* wlane = wp + lane * 16;
    int kr0 = (ho + 1) & 1;
    int kc0 = (par + 1) & 1;

    v8f acc[NBLK] = {};
#pragma unroll
    for (int tr = 0; tr < 2; ++tr) {
        int kr = kr0 + 2 * tr;
        int hi = (ho + 1 - kr) >> 1;
        if (hi >= 0 && hi < Hin) {                    // wave-uniform
#pragma unroll
            for (int tc = 0; tc < 2; ++tc) {
                int kc = kc0 + 2 * tc;
                int wi = (wo + 1 - kc) >> 1;          // per-lane, parity exact
                bool vld = (wi >= 0) && (wi < Win);
                int tap = kr * 4 + kc;                // uniform scalar
                const bf16_t* ap = in + ((size_t)(b * Hin + hi) * Win + (vld ? wi : 0)) * Cin + hv * 8;
#pragma unroll
                for (int q = 0; q < NQ; ++q) {
                    v8bf alo = {}, ahi = {};
                    if (vld) {
                        alo = *(const v8bf*)(ap + q * 32);
                        ahi = *(const v8bf*)(ap + q * 32 + 16);
                    }
                    v16bf a = __builtin_shufflevector(alo, ahi,
                                0,1,2,3,4,5,6,7,8,9,10,11,12,13,14,15);
#pragma unroll
                    for (int j = 0; j < NBLK; ++j) {
                        v16bf bb = *(const v16bf*)(wlane + ((tap * NQ + q) * NBC + j) * 512);
                        acc[j] = __builtin_amdgcn_wmma_f32_16x16x32_bf16(
                                    false, a, false, bb, (short)0, acc[j], false, false);
                    }
                }
            }
        }
    }

#pragma unroll
    for (int j = 0; j < NBLK; ++j) {
        int n = j * 16 + (lane & 15);
        float bz = bias[n];
#pragma unroll
        for (int r = 0; r < 8; ++r) {
            int pix = wbase + 2 * (hv * 8 + r);
            out[((size_t)(b * Ho + ho) * Wo + pix) * Cout + n] = (bf16_t)(acc[j][r] + bz);
        }
    }
}

// ---------------------------------------------------------------------------
// Deformable conv (ks=3, pad=1, C=64): bilinear gather fused into A-fragment
// construction; K = C per tap, 9 taps. A fragment (expensive gather) reused
// by 4 WMMAs -> full 64 output channels per wave.
// ---------------------------------------------------------------------------
__global__ __launch_bounds__(32)
void dconv_wmma(const bf16_t* __restrict__ x, const float* __restrict__ off,
                const bf16_t* __restrict__ wp, const float* __restrict__ bias,
                bf16_t* __restrict__ out, int H, int W) {
    constexpr int NQ = 2, NBLK = 4, NBC = 4, C = 64;
    int lane = threadIdx.x;
    int m  = lane & 15;
    int hv = lane >> 4;
    int w0 = blockIdx.x * 16;
    int h  = blockIdx.y;
    int b  = blockIdx.z;
    int w = w0 + m;

    const bf16_t* wlane = wp + lane * 16;
    const float* offp = off + ((size_t)(b * H + h) * W + w) * 32;

    v8f acc[NBLK] = {};
#pragma unroll
    for (int tap = 0; tap < 9; ++tap) {
        float offr = offp[tap];
        float offc = offp[9 + tap];
        float pr = (float)(h + 1) + (float)(tap / 3 - 1) + offr;
        float pc = (float)(w + 1) + (float)(tap % 3 - 1) + offc;
        pr = fminf(fmaxf(pr, 0.f), (float)(H + 1));
        pc = fminf(fmaxf(pc, 0.f), (float)(W + 1));
        float r0f = floorf(pr), c0f = floorf(pc);
        float fr = pr - r0f, fc = pc - c0f;
        int r0 = (int)r0f, c0 = (int)c0f;
        int r1 = min(r0 + 1, H + 1), c1 = min(c0 + 1, W + 1);
        float w00 = (1.f - fr) * (1.f - fc), w01 = (1.f - fr) * fc;
        float w10 = fr * (1.f - fc),         w11 = fr * fc;
        // padded coords -> original image: valid iff 1 <= r <= H, 1 <= c <= W
        bool v00 = (r0 >= 1) && (r0 <= H) && (c0 >= 1) && (c0 <= W);
        bool v01 = (r0 >= 1) && (r0 <= H) && (c1 >= 1) && (c1 <= W);
        bool v10 = (r1 >= 1) && (r1 <= H) && (c0 >= 1) && (c0 <= W);
        bool v11 = (r1 >= 1) && (r1 <= H) && (c1 >= 1) && (c1 <= W);
        const bf16_t* p00 = x + ((size_t)(b * H + (r0 - 1)) * W + (c0 - 1)) * C;
        const bf16_t* p01 = x + ((size_t)(b * H + (r0 - 1)) * W + (c1 - 1)) * C;
        const bf16_t* p10 = x + ((size_t)(b * H + (r1 - 1)) * W + (c0 - 1)) * C;
        const bf16_t* p11 = x + ((size_t)(b * H + (r1 - 1)) * W + (c1 - 1)) * C;

#pragma unroll
        for (int q = 0; q < NQ; ++q) {
            int k0 = q * 32 + hv * 8;
            float tmp[16];
#pragma unroll
            for (int run = 0; run < 2; ++run) {
                int ofs = k0 + run * 16;
                v8bf s00 = {}, s01 = {}, s10 = {}, s11 = {};
                if (v00) s00 = *(const v8bf*)(p00 + ofs);
                if (v01) s01 = *(const v8bf*)(p01 + ofs);
                if (v10) s10 = *(const v8bf*)(p10 + ofs);
                if (v11) s11 = *(const v8bf*)(p11 + ofs);
#pragma unroll
                for (int e = 0; e < 8; ++e)
                    tmp[run * 8 + e] = w00 * (float)s00[e] + w01 * (float)s01[e] +
                                       w10 * (float)s10[e] + w11 * (float)s11[e];
            }
            v16bf a;
#pragma unroll
            for (int e = 0; e < 16; ++e) a[e] = (bf16_t)tmp[e];
#pragma unroll
            for (int j = 0; j < NBLK; ++j) {
                v16bf bb = *(const v16bf*)(wlane + ((tap * NQ + q) * NBC + j) * 512);
                acc[j] = __builtin_amdgcn_wmma_f32_16x16x32_bf16(
                            false, a, false, bb, (short)0, acc[j], false, false);
            }
        }
    }

#pragma unroll
    for (int j = 0; j < NBLK; ++j) {
        int n = j * 16 + (lane & 15);
        float bz = bias[n];
#pragma unroll
        for (int r = 0; r < 8; ++r) {
            int pix = w0 + hv * 8 + r;
            out[((size_t)(b * H + h) * W + pix) * C + n] = (bf16_t)(acc[j][r] + bz);
        }
    }
}

// ---------------------------------------------------------------------------
// Elementwise bf16 add
// ---------------------------------------------------------------------------
__global__ void add_bf16(const bf16_t* __restrict__ a, const bf16_t* __restrict__ b,
                         bf16_t* __restrict__ c, size_t n) {
    size_t i = (size_t)blockIdx.x * 256 + threadIdx.x;
    if (i < n) c[i] = (bf16_t)((float)a[i] + (float)b[i]);
}

// ---------------------------------------------------------------------------
extern "C" void kernel_launch(void* const* d_in, const int* in_sizes, int n_in,
                              void* d_out, int out_size, void* d_ws, size_t ws_size,
                              hipStream_t stream) {
    const float* dem      = (const float*)d_in[0];
    const float* rs       = (const float*)d_in[1];
    const float* w_down1  = (const float*)d_in[2];  const float* b_down1  = (const float*)d_in[3];
    const float* w_c1     = (const float*)d_in[4];  const float* b_c1     = (const float*)d_in[5];
    const float* w_c12    = (const float*)d_in[6];  const float* b_c12    = (const float*)d_in[7];
    const float* w_up1    = (const float*)d_in[8];  const float* b_up1    = (const float*)d_in[9];
    const float* w_downrs = (const float*)d_in[10]; const float* b_downrs = (const float*)d_in[11];
    const float* w_off    = (const float*)d_in[12]; const float* b_off    = (const float*)d_in[13];
    const float* w_c3     = (const float*)d_in[14]; const float* b_c3     = (const float*)d_in[15];
    const float* w_d1o    = (const float*)d_in[16]; const float* b_d1o    = (const float*)d_in[17];
    const float* w_d1     = (const float*)d_in[18]; const float* b_d1     = (const float*)d_in[19];
    const float* w_d2o    = (const float*)d_in[20]; const float* b_d2o    = (const float*)d_in[21];
    const float* w_d2     = (const float*)d_in[22]; const float* b_d2     = (const float*)d_in[23];
    const float* w_up2    = (const float*)d_in[24]; const float* b_up2    = (const float*)d_in[25];

    const int B = 2, C = 64, H = 256, W = 256, H2 = 128, W2 = 128;
    const size_t Mi = 1u << 20;
    char* ws = (char*)d_ws;

    // workspace schedule (peak ~98 MiB, bf16 NHWC unless noted)
    bf16_t* TCAT = (bf16_t*)(ws + 0);          // [0,32Mi)  concat(dem,rs) 128ch@256^2
    bf16_t* SL0A = (bf16_t*)(ws + 0);          // reuse: fea_2_1a / fea / fea_2_1b
    bf16_t* SL0B = (bf16_t*)(ws + 16 * Mi);    // reuse: sum1 / off1(f32) / sum2
    float*  OFF1 = (float*)(ws + 16 * Mi);
    bf16_t* RSBF = (bf16_t*)(ws + 32 * Mi);    // rs bf16 64ch@256^2
    bf16_t* S2   = (bf16_t*)(ws + 48 * Mi);    // fea_2a 128ch@128^2 ; later rs_2 64ch
    float*  OFF2 = (float*)(ws + 52 * Mi);     // off2 f32 32ch@128^2
    bf16_t* S3b  = (bf16_t*)(ws + 56 * Mi);    // fea_2b ; later fea_2c 64ch@128^2
    bf16_t* S4b  = (bf16_t*)(ws + 60 * Mi);    // fea_1b 64ch@256^2
    bf16_t* S5b  = (bf16_t*)(ws + 76 * Mi);    // fea_1_1 64ch@256^2
    bf16_t* PW[12];
    for (int i = 0; i < 12; ++i) PW[i] = (bf16_t*)(ws + 92 * Mi + (size_t)i * 512 * 1024);

    // --- layout conversions -------------------------------------------------
    {
        size_t tot = (size_t)B * C * H * W;
        int blk = (int)((tot + 255) / 256);
        nchw_to_nhwc_bf16<<<blk, 256, 0, stream>>>(dem, TCAT, B, C, H, W, 128, 0);
        nchw_to_nhwc_bf16<<<blk, 256, 0, stream>>>(rs,  TCAT, B, C, H, W, 128, 64);
        nchw_to_nhwc_bf16<<<blk, 256, 0, stream>>>(rs,  RSBF, B, C, H, W, 64, 0);
    }

    // --- weight prepacks ----------------------------------------------------
    auto pack = [&](const float* src, bf16_t* dst, int Cout, int Cin, int ks,
                    int nq, int nbc, int mode) {
        int total = ks * ks * nq * nbc * 512;
        prepack_w<<<(total + 255) / 256, 256, 0, stream>>>(src, dst, Cout, Cin, ks, nq, nbc, mode);
    };
    pack(w_down1,  PW[0], 128, 128, 3, 4, 8, 0);
    pack(w_c1,     PW[1],  64, 128, 3, 4, 4, 0);
    pack(w_c12,    PW[2],  64, 128, 3, 4, 4, 0);
    pack(w_up1,    PW[3],  64,  64, 4, 2, 4, 1);
    pack(w_downrs, PW[4],  64,  64, 3, 2, 4, 0);
    pack(w_off,    PW[5],  64,  64, 1, 2, 4, 0);
    pack(w_c3,     PW[6],  64,  64, 1, 2, 4, 0);
    pack(w_d1o,    PW[7],  18,  64, 3, 2, 2, 0);
    pack(w_d1,     PW[8],  64,  64, 3, 2, 4, 0);
    pack(w_d2o,    PW[9],  18,  64, 3, 2, 2, 0);
    pack(w_d2,     PW[10], 64,  64, 3, 2, 4, 0);
    pack(w_up2,    PW[11], 64,  64, 4, 2, 4, 1);

    // --- network ------------------------------------------------------------
    // fea_2a = conv(cat, w_down1, s2)          128ch @128^2
    conv_wmma<0,4,4,3,8><<<dim3(W2/16, H2, B*2), 32, 0, stream>>>(
        TCAT, PW[0], b_down1, S2, H, W, H2, W2, 128, 128, 2, 1);
    // fea_1b = conv(cat, w_c1, s1)             64ch @256^2
    conv_wmma<0,4,4,3,4><<<dim3(W/16, H, B), 32, 0, stream>>>(
        TCAT, PW[1], b_c1, S4b, H, W, H, W, 64, 64, 1, 1);
    // fea_2b = conv(fea_2a, w_c12, s1)         64ch @128^2
    conv_wmma<0,4,4,3,4><<<dim3(W2/16, H2, B), 32, 0, stream>>>(
        S2, PW[2], b_c12, S3b, H2, W2, H2, W2, 64, 64, 1, 1);
    // fea_2_1a = convT(fea_2b, w_up1)          64ch @256^2  -> SL0A
    convt_wmma<<<dim3(W/32, H, B*2), 32, 0, stream>>>(
        S3b, PW[3], b_up1, SL0A, H2, W2, H, W);
    // sum1 = fea_1b + fea_2_1a                 -> SL0B
    {
        size_t n = (size_t)B * H * W * 64;
        add_bf16<<<(int)((n + 255) / 256), 256, 0, stream>>>(S4b, SL0A, SL0B, n);
    }
    // fea = conv1x1(sum1, w_off)               -> SL0A
    conv_wmma<0,2,4,1,4><<<dim3(W/16, H, B), 32, 0, stream>>>(
        SL0B, PW[5], b_off, SL0A, H, W, H, W, 64, 64, 1, 0);
    // off1 = conv3x3(fea, w_d1o)  fp32 NHWC-32 -> OFF1 (SL0B region)
    conv_wmma<1,2,2,3,2><<<dim3(W/16, H, B), 32, 0, stream>>>(
        SL0A, PW[7], b_d1o, OFF1, H, W, H, W, 18, 32, 1, 1);
    // fea_1_1 = deform(rs, off1, w_d1)         -> S5b
    dconv_wmma<<<dim3(W/16, H, B), 32, 0, stream>>>(
        RSBF, OFF1, PW[8], b_d1, S5b, H, W);
    // rs_2 = conv(rs, w_downrs, s2)            64ch @128^2 -> S2
    conv_wmma<0,2,4,3,4><<<dim3(W2/16, H2, B), 32, 0, stream>>>(
        RSBF, PW[4], b_downrs, S2, H, W, H2, W2, 64, 64, 2, 1);
    // off2 = conv3x3(fea_2b, w_d2o) fp32       -> OFF2
    conv_wmma<1,2,2,3,2><<<dim3(W2/16, H2, B), 32, 0, stream>>>(
        S3b, PW[9], b_d2o, OFF2, H2, W2, H2, W2, 18, 32, 1, 1);
    // fea_2c = deform(rs_2, off2, w_d2)        -> S3b (fea_2b dead)
    dconv_wmma<<<dim3(W2/16, H2, B), 32, 0, stream>>>(
        S2, OFF2, PW[10], b_d2, S3b, H2, W2);
    // fea_2_1b = convT(fea_2c, w_up2)          -> SL0A
    convt_wmma<<<dim3(W/32, H, B*2), 32, 0, stream>>>(
        S3b, PW[11], b_up2, SL0A, H2, W2, H, W);
    // sum2 = fea_1_1 + fea_2_1b                -> SL0B
    {
        size_t n = (size_t)B * H * W * 64;
        add_bf16<<<(int)((n + 255) / 256), 256, 0, stream>>>(S5b, SL0A, SL0B, n);
    }
    // out = conv1x1(sum2, w_c3)  fp32 NCHW     -> d_out
    conv_wmma<2,2,4,1,4><<<dim3(W/16, H, B), 32, 0, stream>>>(
        SL0B, PW[6], b_c3, d_out, H, W, H, W, 64, 64, 1, 0);
}
